// EGINConv_56599079027144
// MI455X (gfx1250) — matrix operand order repaired and checked by reading, not compiled, not run on hardware
//
#include <hip/hip_runtime.h>
#include <hip/hip_bf16.h>

#define D      128
#define TWO_D  256
#define NNODES 50000
#define NEDGES 600000

typedef __attribute__((ext_vector_type(16))) _Float16 v16h;
typedef __attribute__((ext_vector_type(8)))  _Float16 v8h;
typedef __attribute__((ext_vector_type(8)))  float    v8f;

// Native CDNA5 no-return f32 atomic add at device scope (L2 RMW units).
// Tracked by STOREcnt; S_ENDPGM does an implicit wait-idle, and stream-order
// kernel boundaries give cross-kernel visibility.
__device__ __forceinline__ void atomic_add_f32(float* p, float v) {
    asm volatile("global_atomic_add_f32 %0, %1, off scope:SCOPE_DEV"
                 :: "v"(p), "v"(v) : "memory");
}

// ---------------- zero workspace (new_x + BN stat counters) ----------------
__global__ void zero_ws_kernel(float* __restrict__ p, int n) {
    int i = blockIdx.x * blockDim.x + threadIdx.x;
    if (i < n) p[i] = 0.0f;
}

// ---------------- convert + swizzle W1/W2 fp32 -> f16 B-fragments ----------
// Packed layout: fragment (ct, kkidx) holds lane-major data; lane's 16 halfs
// contiguous:  Wp[((ct*KB + kkidx)*32 + lane)*16 + j] = W[k][col]
//   with k = kkidx*32 + (lane>>4)*16 + j,  col = ct*16 + (lane&15)
// (contiguous k order, same as the A fragments: lanes 0-15 K=0..15, lanes 16-31 K=16..31)
__global__ void cvt_w_kernel(const float* __restrict__ W1,
                             const float* __restrict__ W2,
                             _Float16* __restrict__ W1p,
                             _Float16* __restrict__ W2p) {
    int t = blockIdx.x * blockDim.x + threadIdx.x;   // 0 .. 4095
    int lane = t & 31;
    int koB  = (lane >> 4) * 16;
    int n    = lane & 15;
    if (t < 2048) {                 // W1: K=128 (KB=4 blocks), N=256 (16 tiles)
        int fid   = t >> 5;         // 0..63
        int ct    = fid >> 2;
        int kkidx = fid & 3;
        #pragma unroll
        for (int j = 0; j < 16; ++j) {
            int k = kkidx * 32 + koB + j;
            W1p[(size_t)t * 16 + j] = (_Float16)W1[k * TWO_D + ct * 16 + n];
        }
    } else if (t < 4096) {          // W2: K=256 (KB=8 blocks), N=128 (8 tiles)
        int t2    = t - 2048;
        int fid   = t2 >> 5;        // 0..63
        int ct    = fid >> 3;
        int kkidx = fid & 7;
        #pragma unroll
        for (int j = 0; j < 16; ++j) {
            int k = kkidx * 32 + koB + j;
            W2p[(size_t)t2 * 16 + j] = (_Float16)W2[k * D + ct * 16 + n];
        }
    }
}

// Load a 16-half A fragment from LDS as two aligned 16-byte vectors.
__device__ __forceinline__ v16h load_a_frag(const _Float16* rowp, int kbase) {
    v8h lo = *(const v8h*)(rowp + kbase);
    v8h hi = *(const v8h*)(rowp + kbase + 8);
    return __builtin_shufflevector(lo, hi, 0, 1, 2, 3, 4, 5, 6, 7,
                                   8, 9, 10, 11, 12, 13, 14, 15);
}

// ---------------- edge phase: gather + relu + scatter-add ------------------
// one 32-lane wave per edge, float4 per lane (32*4 = 128 = D)
__global__ __launch_bounds__(256) void edge_scatter_kernel(
    const float* __restrict__ x,
    const int*   __restrict__ ex,
    const int*   __restrict__ src,
    const int*   __restrict__ dst,
    const float* __restrict__ e0,
    const float* __restrict__ e1,
    const float* __restrict__ e2,
    float*       __restrict__ newx)
{
    int gtid = blockIdx.x * blockDim.x + threadIdx.x;
    int e = gtid >> 5;
    if (e >= NEDGES) return;
    int lane = gtid & 31;
    int c = lane << 2;

    int i0 = ex[3 * e + 0];
    int i1 = ex[3 * e + 1];
    int i2 = ex[3 * e + 2];
    int s  = src[e];
    int d  = dst[e];

    const float4 xv = *reinterpret_cast<const float4*>(x  + (size_t)s  * D + c);
    const float4 a0 = *reinterpret_cast<const float4*>(e0 + (size_t)i0 * D + c);
    const float4 a1 = *reinterpret_cast<const float4*>(e1 + (size_t)i1 * D + c);
    const float4 a2 = *reinterpret_cast<const float4*>(e2 + (size_t)i2 * D + c);

    float m0 = fmaxf(xv.x + a0.x + a1.x + a2.x, 0.0f);
    float m1 = fmaxf(xv.y + a0.y + a1.y + a2.y, 0.0f);
    float m2 = fmaxf(xv.z + a0.z + a1.z + a2.z, 0.0f);
    float m3 = fmaxf(xv.w + a0.w + a1.w + a2.w, 0.0f);

    float* o = newx + (size_t)d * D + c;
    atomic_add_f32(o + 0, m0);
    atomic_add_f32(o + 1, m1);
    atomic_add_f32(o + 2, m2);
    atomic_add_f32(o + 3, m3);
}

// ---------------- GEMM1: t = ((1+eps)*x + new_x) @ W1 + b1, + BN stats -----
// block = 256 threads = 8 waves; block tile = 16 rows x 256 cols;
// each wave computes two 16x16 output tiles (cols wave*16 and (wave+8)*16).
__global__ __launch_bounds__(256) void gemm1_bn_kernel(
    const float*    __restrict__ x,
    const float*    __restrict__ newx,
    const float*    __restrict__ epsp,
    const _Float16* __restrict__ W1p,
    const float*    __restrict__ b1p,
    float*          __restrict__ tbuf,
    float*          __restrict__ sumv,
    float*          __restrict__ sumsq)
{
    __shared__ _Float16 Ah[16][D + 8];   // row stride 272 B (16B-aligned), no conflicts

    const int tid  = threadIdx.x;
    const int row0 = blockIdx.x * 16;
    const float ep1 = 1.0f + epsp[0];

    for (int i = tid; i < 16 * D; i += 256) {
        int r = i >> 7, c = i & (D - 1);
        size_t idx = (size_t)(row0 + r) * D + c;
        Ah[r][c] = (_Float16)(ep1 * x[idx] + newx[idx]);
    }
    __syncthreads();

    const int wave = tid >> 5;
    const int lane = tid & 31;
    const int m    = lane & 15;          // A row
    const int khA  = (lane >> 4) * 16;   // A: K sub-offset (contiguous, matches packed B)
    const int n    = lane & 15;          // B/C column within tile

    const int ct0 = wave, ct1 = wave + 8;
    const _Float16* rowp = &Ah[m][0];
    v8f acc0 = {}; v8f acc1 = {};

    #pragma unroll
    for (int kkidx = 0; kkidx < 4; ++kkidx) {
        v16h a = load_a_frag(rowp, kkidx * 32 + khA);
        v16h b0 = *(const v16h*)(W1p + ((size_t)((ct0 * 4 + kkidx) * 32 + lane) * 16));
        v16h b1 = *(const v16h*)(W1p + ((size_t)((ct1 * 4 + kkidx) * 32 + lane) * 16));
        acc0 = __builtin_amdgcn_wmma_f32_16x16x32_f16(false, a, false, b0,
                                                      (short)0, acc0, false, false);
        acc1 = __builtin_amdgcn_wmma_f32_16x16x32_f16(false, a, false, b1,
                                                      (short)0, acc1, false, false);
    }

    // epilogue: +bias, store t, per-column sum/sumsq
    // (C/D layout: lane<16 -> rows 0-7, lane>=16 -> rows 8-15, col = lane&15)
    const int mbase = (lane >> 4) * 8;
    #pragma unroll
    for (int tsel = 0; tsel < 2; ++tsel) {
        const int col = (tsel == 0 ? ct0 : ct1) * 16 + n;
        v8f acc = (tsel == 0) ? acc0 : acc1;
        const float bias = b1p[col];
        float ls = 0.0f, ls2 = 0.0f;
        #pragma unroll
        for (int r = 0; r < 8; ++r) {
            float v = acc[r] + bias;
            tbuf[(size_t)(row0 + mbase + r) * TWO_D + col] = v;
            ls  += v;
            ls2 += v * v;
        }
        atomic_add_f32(&sumv[col],  ls);
        atomic_add_f32(&sumsq[col], ls2);
    }
}

// ---------------- BN finalize + ReLU + GEMM2: out = relu(bn(t)) @ W2 + b2 --
// block tile = 16 rows x 128 cols; 8 waves, one 16x16 tile each; K = 256.
__global__ __launch_bounds__(256) void bn_relu_gemm2_kernel(
    const float*    __restrict__ tbuf,
    const float*    __restrict__ sumv,
    const float*    __restrict__ sumsq,
    const float*    __restrict__ gamma,
    const float*    __restrict__ beta,
    const _Float16* __restrict__ W2p,
    const float*    __restrict__ b2p,
    float*          __restrict__ out)
{
    __shared__ _Float16 Uh[16][TWO_D + 8];   // row stride 528 B (16B-aligned)
    __shared__ float sscale[TWO_D];
    __shared__ float sshift[TWO_D];

    const int tid  = threadIdx.x;
    const int row0 = blockIdx.x * 16;

    if (tid < TWO_D) {
        float mu  = sumv[tid]  * (1.0f / (float)NNODES);
        float var = sumsq[tid] * (1.0f / (float)NNODES) - mu * mu;
        float rs  = rsqrtf(var + 1e-5f);
        float sc  = gamma[tid] * rs;
        sscale[tid] = sc;
        sshift[tid] = beta[tid] - mu * sc;
    }
    __syncthreads();

    for (int i = tid; i < 16 * TWO_D; i += 256) {
        int r = i >> 8, c = i & (TWO_D - 1);
        float v = tbuf[(size_t)(row0 + r) * TWO_D + c];
        v = fmaf(v, sscale[c], sshift[c]);
        Uh[r][c] = (_Float16)fmaxf(v, 0.0f);
    }
    __syncthreads();

    const int wave = tid >> 5;
    const int lane = tid & 31;
    const int m    = lane & 15;
    const int khA  = (lane >> 4) * 16;
    const int n    = lane & 15;

    const _Float16* rowp = &Uh[m][0];
    v8f acc = {};
    #pragma unroll
    for (int kkidx = 0; kkidx < 8; ++kkidx) {
        v16h a = load_a_frag(rowp, kkidx * 32 + khA);
        v16h b = *(const v16h*)(W2p + ((size_t)((wave * 8 + kkidx) * 32 + lane) * 16));
        acc = __builtin_amdgcn_wmma_f32_16x16x32_f16(false, a, false, b,
                                                     (short)0, acc, false, false);
    }

    const int mbase = (lane >> 4) * 8;
    const int col   = wave * 16 + n;
    const float bias = b2p[col];
    #pragma unroll
    for (int r = 0; r < 8; ++r)
        out[(size_t)(row0 + mbase + r) * D + col] = acc[r] + bias;
}

// ---------------------------------------------------------------------------
extern "C" void kernel_launch(void* const* d_in, const int* in_sizes, int n_in,
                              void* d_out, int out_size, void* d_ws, size_t ws_size,
                              hipStream_t stream) {
    const float* x     = (const float*)d_in[0];
    const int*   ex    = (const int*)  d_in[1];
    const int*   src   = (const int*)  d_in[2];
    const int*   dst   = (const int*)  d_in[3];
    const float* e0    = (const float*)d_in[4];
    const float* e1    = (const float*)d_in[5];
    const float* e2    = (const float*)d_in[6];
    const float* eps   = (const float*)d_in[7];
    const float* W1    = (const float*)d_in[8];
    const float* b1    = (const float*)d_in[9];
    const float* gamma = (const float*)d_in[10];
    const float* beta  = (const float*)d_in[11];
    const float* W2    = (const float*)d_in[12];
    const float* b2    = (const float*)d_in[13];
    float* out = (float*)d_out;

    // workspace: [ new_x (N*D) | sum (256) | sumsq (256) | t (N*2D) | W1p | W2p ]
    float* newx  = (float*)d_ws;
    float* sumv  = newx + (size_t)NNODES * D;
    float* sumsq = sumv + TWO_D;
    float* tbuf  = sumsq + TWO_D;
    _Float16* W1p = (_Float16*)(tbuf + (size_t)NNODES * TWO_D);
    _Float16* W2p = W1p + D * TWO_D;

    const int zn = NNODES * D + 2 * TWO_D;
    zero_ws_kernel<<<(zn + 255) / 256, 256, 0, stream>>>(newx, zn);
    cvt_w_kernel<<<(4096 + 255) / 256, 256, 0, stream>>>(W1, W2, W1p, W2p);
    edge_scatter_kernel<<<(NEDGES * 32 + 255) / 256, 256, 0, stream>>>(
        x, ex, src, dst, e0, e1, e2, newx);
    gemm1_bn_kernel<<<NNODES / 16, 256, 0, stream>>>(
        x, newx, eps, W1p, b1, tbuf, sumv, sumsq);
    bn_relu_gemm2_kernel<<<NNODES / 16, 256, 0, stream>>>(
        tbuf, sumv, sumsq, gamma, beta, W2p, b2, out);
}
